// Q_network_71846212927505
// MI455X (gfx1250) — matrix operand order repaired and checked by reading, not compiled, not run on hardware
//
#include <hip/hip_runtime.h>
#include <hip/hip_bf16.h>

typedef _Float16 f16;
typedef __attribute__((ext_vector_type(16))) _Float16 v16h;
typedef __attribute__((ext_vector_type(8)))  float    v8f;

#define Bc  32
#define Tc  128
#define Nc  32
#define Ec  64
#define Hc  4
#define DKc 16
#define NEc 16
#define ADc 64
#define EDc 32
#define BTc (Bc*Tc)       /* 4096   */
#define ROWS (BTc*Nc)     /* 131072 */

__device__ __forceinline__ float geluf(float x) {
  return 0.5f * x * (1.0f + erff(x * 0.70710678118654752440f));
}
__device__ __forceinline__ float sigm(float x) { return 1.0f / (1.0f + expf(-x)); }

// ---- CDNA5 WMMA fragment helpers (ISA 7.12.2 layouts, wave32) -------------
// 16-bit A 16x32 (MxK): lane L holds M=L%16; element e -> K per spec.
__device__ __forceinline__ int a_k_of(int lane, int e) {
  int g = (lane >> 4) & 1;
  int v = e >> 1;
  return ((v < 4) ? 0 : 16) + g * 8 + (v & 3) * 2 + (e & 1);
}
template <typename F>
__device__ __forceinline__ v16h make_frag_a(int lane, F f) {
  v16h a;
  int m = lane & 15;
#pragma unroll
  for (int e = 0; e < 16; ++e) a[e] = (f16)f(m, a_k_of(lane, e));
  return a;
}
// 16-bit B 32x16 (KxN): lane L holds N=L%16; K = (e<8 ? 0 : 16) + (L/16)*8 + (e&7).
template <typename F>
__device__ __forceinline__ v16h make_frag_b(int lane, F f) {
  v16h b;
  int n = lane & 15, g = (lane >> 4) & 1;
#pragma unroll
  for (int e = 0; e < 16; ++e) b[e] = (f16)f(((e < 8) ? 0 : 16) + g * 8 + (e & 7), n);
  return b;
}
// f32 C/D 16x16: lane holds col=L%16, rows i + 8*(L/16).
template <typename F>
__device__ __forceinline__ void store_c(int lane, const v8f& c, F f) {
  int col = lane & 15, g = (lane >> 4) & 1;
#pragma unroll
  for (int i = 0; i < 8; ++i) f(i + 8 * g, col, c[i]);
}
__device__ __forceinline__ v8f wmma(v16h a, v16h b, v8f c) {
  return __builtin_amdgcn_wmma_f32_16x16x32_f16(false, a, false, b, (short)0, c,
                                                false, false);
}

// ---- K1: enemy embedding + reduction over NE --------------------------------
__global__ void k_enemy(const float* __restrict__ es, const float* __restrict__ W_enemy,
                        const float* __restrict__ b_enemy, const float* __restrict__ enemy_emb,
                        float* __restrict__ esum) {
  int bt = blockIdx.x;
  int lane = threadIdx.x;
  const float* E = es + (size_t)bt * NEc * EDc;
  v16h a = make_frag_a(lane, [&](int m, int k) { return E[m * EDc + k]; });
  float colsum[4];
#pragma unroll
  for (int t = 0; t < 4; ++t) {
    v16h b = make_frag_b(lane, [&](int k, int n) { return W_enemy[k * Ec + t * 16 + n]; });
    v8f c = {};
    c = wmma(a, b, c);
    int col = t * 16 + (lane & 15);
    float p = 0.f;
#pragma unroll
    for (int i = 0; i < 8; ++i) p += geluf(c[i] + b_enemy[col]);
    p += __shfl_xor(p, 16, 32);  // combine rows 0-7 with rows 8-15
    colsum[t] = p;
  }
  if (lane < 16) {
#pragma unroll
    for (int t = 0; t < 4; ++t) {
      int col = t * 16 + lane;
      float ee = 0.f;
      for (int j = 0; j < NEc; ++j) ee += enemy_emb[j * Ec + col];
      esum[(size_t)bt * Ec + col] = colsum[t] + ee;
    }
  }
}

// ---- K2: ally embedding + agent_emb + esum, LayerNorm1, oae -----------------
__global__ void k_se(const float* __restrict__ states, const float* __restrict__ W_ally,
                     const float* __restrict__ b_ally, const float* __restrict__ agent_emb,
                     const float* __restrict__ esum, const float* __restrict__ ln1_g,
                     const float* __restrict__ ln1_b, const int* __restrict__ actions,
                     const float* __restrict__ action_emb, float* __restrict__ se,
                     float* __restrict__ oae) {
  int R = blockIdx.x * 16;
  int lane = threadIdx.x;
  int bt = R / Nc, n0 = R % Nc;
  __shared__ float buf[16][Ec];
  v16h a0 = make_frag_a(lane, [&](int m, int k) { return states[(size_t)(R + m) * ADc + k]; });
  v16h a1 = make_frag_a(lane, [&](int m, int k) { return states[(size_t)(R + m) * ADc + 32 + k]; });
#pragma unroll
  for (int t = 0; t < 4; ++t) {
    v16h b0 = make_frag_b(lane, [&](int k, int n) { return W_ally[k * Ec + t * 16 + n]; });
    v16h b1 = make_frag_b(lane, [&](int k, int n) { return W_ally[(32 + k) * Ec + t * 16 + n]; });
    v8f c = {};
    c = wmma(a0, b0, c);
    c = wmma(a1, b1, c);
    store_c(lane, c, [&](int m, int col0, float val) {
      int col = t * 16 + col0;
      buf[m][col] = geluf(val + b_ally[col]) + agent_emb[(n0 + m) * Ec + col] +
                    esum[(size_t)bt * Ec + col];
    });
  }
  __syncthreads();
  if (lane < 16) {
    int m = lane;
    float mu = 0.f;
    for (int c = 0; c < Ec; ++c) mu += buf[m][c];
    mu *= (1.f / Ec);
    float var = 0.f;
    for (int c = 0; c < Ec; ++c) { float d = buf[m][c] - mu; var += d * d; }
    var *= (1.f / Ec);
    float rs = rsqrtf(var + 1e-5f);
    int act = actions[R + m];
    for (int c = 0; c < Ec; ++c) {
      float y = (buf[m][c] - mu) * rs * ln1_g[c] + ln1_b[c];
      se[(size_t)(R + m) * Ec + c] = y;
      oae[(size_t)(R + m) * Ec + c] = y + action_emb[act * Ec + c];
    }
  }
}

// ---- K3: generic [rows,64]@[64,64]+bias projection (f16 output) -------------
__global__ void k_proj(const float* __restrict__ in, const float* __restrict__ W,
                       const float* __restrict__ bias, f16* __restrict__ out) {
  int R = blockIdx.x * 16;
  int lane = threadIdx.x;
  v16h a0 = make_frag_a(lane, [&](int m, int k) { return in[(size_t)(R + m) * Ec + k]; });
  v16h a1 = make_frag_a(lane, [&](int m, int k) { return in[(size_t)(R + m) * Ec + 32 + k]; });
#pragma unroll
  for (int t = 0; t < 4; ++t) {
    v16h b0 = make_frag_b(lane, [&](int k, int n) { return W[k * Ec + t * 16 + n]; });
    v16h b1 = make_frag_b(lane, [&](int k, int n) { return W[(32 + k) * Ec + t * 16 + n]; });
    v8f c = {};
    c = wmma(a0, b0, c);
    c = wmma(a1, b1, c);
    store_c(lane, c, [&](int m, int col0, float val) {
      int col = t * 16 + col0;
      out[(size_t)(R + m) * Ec + col] = (f16)(val + bias[col]);
    });
  }
}

// ---- K4: attention per (bt,h): gumbel hard mask, dual softmax, prd, agg -----
__global__ void k_attn(const f16* __restrict__ qg, const f16* __restrict__ kg,
                       const f16* __restrict__ vg, const float* __restrict__ Wh,
                       const float* __restrict__ bh, const float* __restrict__ gumbel,
                       const float* __restrict__ amask, float* __restrict__ agg,
                       float* __restrict__ prd_out) {
  const float MV = -3.402823466e38f;
  int id = blockIdx.x;  // bt*H + h
  int bt = id / Hc, h = id % Hc;
  int lane = threadIdx.x;  // row i (or col j)
  __shared__ float qs[Nc][DKc], ks[Nc][DKc], vs[Nc][DKc];
  __shared__ float sc[Nc][Nc], at[Nc][Nc + 1], fw[Nc][Nc + 1];
  __shared__ float lk2[Nc][2], d1s[Nc], d2s[Nc], am[Nc];
  for (int d = 0; d < DKc; ++d) {
    size_t base = ((size_t)bt * Nc + lane) * Ec + h * DKc + d;
    qs[lane][d] = (float)qg[base];
    ks[lane][d] = (float)kg[base];
    vs[lane][d] = (float)vg[base];
  }
  am[lane] = amask[(size_t)bt * Nc + lane];
  __syncthreads();
  float lk0 = 0.f, lk1 = 0.f;
  for (int d = 0; d < DKc; ++d) {
    lk0 += ks[lane][d] * Wh[(DKc + d) * 2 + 0];
    lk1 += ks[lane][d] * Wh[(DKc + d) * 2 + 1];
  }
  lk2[lane][0] = lk0; lk2[lane][1] = lk1;
  __syncthreads();
  float lq0 = 0.f, lq1 = 0.f;
  for (int d = 0; d < DKc; ++d) {
    lq0 += qs[lane][d] * Wh[d * 2 + 0];
    lq1 += qs[lane][d] * Wh[d * 2 + 1];
  }
  const float* gum = gumbel + (size_t)id * Nc * Nc * 2;
  for (int j = 0; j < Nc; ++j) {
    float s = 0.f;
    for (int d = 0; d < DKc; ++d) s += qs[lane][d] * ks[j][d];
    s *= 0.25f;  // 1/sqrt(DK)
    sc[lane][j] = s;
    float l0 = lq0 + lk2[j][0] + bh[0] + gum[((size_t)lane * Nc + j) * 2 + 0];
    float l1 = lq1 + lk2[j][1] + bh[1] + gum[((size_t)lane * Nc + j) * 2 + 1];
    float hw = (lane == j) ? 1.f : ((l1 > l0) ? 1.f : 0.f);
    float m = (am[j] == 0.f || am[lane] == 0.f || lane == j) ? MV : (1.f - am[j]);
    at[lane][j] = m + (1.f - hw) * MV;
  }
  __syncthreads();
  float rmx = -3.4e38f, rmn = 3.4e38f, cmx = -3.4e38f, cmn = 3.4e38f;
  for (int t = 0; t < Nc; ++t) {
    float svr = (at[lane][t] != MV) ? sc[lane][t] : 0.f;
    rmx = fmaxf(rmx, svr); rmn = fminf(rmn, svr);
    float svc = (at[t][lane] != MV) ? sc[t][lane] : 0.f;
    cmx = fmaxf(cmx, svc); cmn = fminf(cmn, svc);
  }
  d1s[lane] = rmx - rmn + 1e-5f;
  d2s[lane] = cmx - cmn + 1e-5f;
  __syncthreads();
  // row softmax -> fw
  {
    float d1 = d1s[lane], mx = -3.4e38f;
    for (int j = 0; j < Nc; ++j) mx = fmaxf(mx, sc[lane][j] / d1 + at[lane][j]);
    float ssum = 0.f;
    for (int j = 0; j < Nc; ++j) ssum += expf(sc[lane][j] / d1 + at[lane][j] - mx);
    float inv = 1.f / ssum;
    for (int j = 0; j < Nc; ++j) {
      float wv = expf(sc[lane][j] / d1 + at[lane][j] - mx) * inv;
      float fv = (lane == j) ? 1.f : wv;
      fw[lane][j] = fv * am[lane] * am[j];
    }
  }
  // column softmax (axis=-2, note den2 indexed by row per reference) -> prd
  {
    float mx = -3.4e38f;
    for (int i = 0; i < Nc; ++i) mx = fmaxf(mx, sc[i][lane] / d2s[i] + at[i][lane]);
    float ssum = 0.f;
    for (int i = 0; i < Nc; ++i) ssum += expf(sc[i][lane] / d2s[i] + at[i][lane] - mx);
    float inv = 1.f / ssum;
    for (int i = 0; i < Nc; ++i) {
      float pv = expf(sc[i][lane] / d2s[i] + at[i][lane] - mx) * inv;
      pv = (i == lane) ? 1.f : pv;
      pv *= am[i] * am[lane];
      prd_out[((size_t)id * Nc + i) * Nc + lane] = pv;
    }
  }
  // agg row (fw row written by this lane, vs loaded pre-barrier)
  for (int d = 0; d < DKc; ++d) {
    float s = 0.f;
    for (int j = 0; j < Nc; ++j) s += fw[lane][j] * vs[j][d];
    agg[((size_t)bt * Nc + lane) * Ec + h * DKc + d] = s;
  }
}

// ---- K5: fused residual + LN2 + FFN + LN3 + gelu(Wc) -> x3 (f16) ------------
__global__ void k_ffn(const float* __restrict__ oae, const float* __restrict__ agg,
                      const float* __restrict__ ln2_g, const float* __restrict__ ln2_b,
                      const float* __restrict__ Wl1, const float* __restrict__ bl1,
                      const float* __restrict__ Wl2, const float* __restrict__ bl2,
                      const float* __restrict__ ln3_g, const float* __restrict__ ln3_b,
                      const float* __restrict__ Wc, const float* __restrict__ bc,
                      f16* __restrict__ x3) {
  int R = blockIdx.x * 16;
  int lane = threadIdx.x;
  __shared__ float x1[16][Ec];
  __shared__ float tmp[16][Ec];
  for (int idx = lane; idx < 16 * Ec; idx += 32) {
    int m = idx >> 6, c = idx & 63;
    tmp[m][c] = oae[(size_t)(R + m) * Ec + c] + agg[(size_t)(R + m) * Ec + c];
  }
  __syncthreads();
  if (lane < 16) {  // LN2 -> x1
    int m = lane;
    float mu = 0.f;
    for (int c = 0; c < Ec; ++c) mu += tmp[m][c];
    mu *= (1.f / Ec);
    float var = 0.f;
    for (int c = 0; c < Ec; ++c) { float d = tmp[m][c] - mu; var += d * d; }
    float rs = rsqrtf(var * (1.f / Ec) + 1e-5f);
    for (int c = 0; c < Ec; ++c) x1[m][c] = (tmp[m][c] - mu) * rs * ln2_g[c] + ln2_b[c];
  }
  __syncthreads();
  // GEMM1: gelu(x1@Wl1+bl1) -> tmp
  {
    v16h a0 = make_frag_a(lane, [&](int m, int k) { return x1[m][k]; });
    v16h a1 = make_frag_a(lane, [&](int m, int k) { return x1[m][32 + k]; });
#pragma unroll
    for (int t = 0; t < 4; ++t) {
      v16h b0 = make_frag_b(lane, [&](int k, int n) { return Wl1[k * Ec + t * 16 + n]; });
      v16h b1 = make_frag_b(lane, [&](int k, int n) { return Wl1[(32 + k) * Ec + t * 16 + n]; });
      v8f c = {};
      c = wmma(a0, b0, c);
      c = wmma(a1, b1, c);
      store_c(lane, c, [&](int m, int col0, float val) {
        int col = t * 16 + col0;
        tmp[m][col] = geluf(val + bl1[col]);
      });
    }
  }
  __syncthreads();
  // GEMM2: x1 + tmp@Wl2+bl2 -> tmp (pre-LN3 sum)
  {
    v16h a0 = make_frag_a(lane, [&](int m, int k) { return tmp[m][k]; });
    v16h a1 = make_frag_a(lane, [&](int m, int k) { return tmp[m][32 + k]; });
    __syncthreads();  // all frag loads of tmp complete before overwriting
#pragma unroll
    for (int t = 0; t < 4; ++t) {
      v16h b0 = make_frag_b(lane, [&](int k, int n) { return Wl2[k * Ec + t * 16 + n]; });
      v16h b1 = make_frag_b(lane, [&](int k, int n) { return Wl2[(32 + k) * Ec + t * 16 + n]; });
      v8f c = {};
      c = wmma(a0, b0, c);
      c = wmma(a1, b1, c);
      store_c(lane, c, [&](int m, int col0, float val) {
        int col = t * 16 + col0;
        tmp[m][col] = x1[m][col] + val + bl2[col];
      });
    }
  }
  __syncthreads();
  if (lane < 16) {  // LN3 in place
    int m = lane;
    float mu = 0.f;
    for (int c = 0; c < Ec; ++c) mu += tmp[m][c];
    mu *= (1.f / Ec);
    float var = 0.f;
    for (int c = 0; c < Ec; ++c) { float d = tmp[m][c] - mu; var += d * d; }
    float rs = rsqrtf(var * (1.f / Ec) + 1e-5f);
    for (int c = 0; c < Ec; ++c) tmp[m][c] = (tmp[m][c] - mu) * rs * ln3_g[c] + ln3_b[c];
  }
  __syncthreads();
  // GEMM3: gelu(x2@Wc+bc) -> x3 (f16)
  {
    v16h a0 = make_frag_a(lane, [&](int m, int k) { return tmp[m][k]; });
    v16h a1 = make_frag_a(lane, [&](int m, int k) { return tmp[m][32 + k]; });
#pragma unroll
    for (int t = 0; t < 4; ++t) {
      v16h b0 = make_frag_b(lane, [&](int k, int n) { return Wc[k * Ec + t * 16 + n]; });
      v16h b1 = make_frag_b(lane, [&](int k, int n) { return Wc[(32 + k) * Ec + t * 16 + n]; });
      v8f c = {};
      c = wmma(a0, b0, c);
      c = wmma(a1, b1, c);
      store_c(lane, c, [&](int m, int col0, float val) {
        int col = t * 16 + col0;
        x3[(size_t)(R + m) * Ec + col] = (f16)geluf(val + bc[col]);
      });
    }
  }
}

// ---- K6: persistent GRU over T steps, weights + h in LDS, gates in C-layout -
__global__ void k_gru(const f16* __restrict__ x3, const float* __restrict__ h0,
                      const float* __restrict__ Wih, const float* __restrict__ Whh,
                      const float* __restrict__ bih, const float* __restrict__ bhh,
                      const float* __restrict__ lnq_g, const float* __restrict__ lnq_b,
                      const float* __restrict__ Wout, const float* __restrict__ bout,
                      float* __restrict__ qv, float* __restrict__ hT) {
  __shared__ f16 WihT[Ec][3 * Ec];  // [64][192] transposed
  __shared__ f16 WhhT[Ec][3 * Ec];
  __shared__ float hS[2][16][Ec];
  int tid = threadIdx.x, lane = tid & 31, w = tid >> 5;
  for (int idx = tid; idx < 3 * Ec * Ec; idx += blockDim.x) {
    int n = idx / Ec, k2 = idx % Ec;
    WihT[k2][n] = (f16)Wih[n * Ec + k2];
    WhhT[k2][n] = (f16)Whh[n * Ec + k2];
  }
  int Rw = blockIdx.x * 32 + w * 16;  // rows Rw..Rw+16 of the 1024 agent rows
  for (int idx = lane; idx < 16 * Ec; idx += 32) {
    int m = idx >> 6, c = idx & 63;
    hS[w][m][c] = h0[(size_t)(Rw + m) * Ec + c];
  }
  __syncthreads();
  int b = Rw >> 5;        // all 16 rows share the same batch index
  int nbase = Rw & 31;
  float bo = bout[0];
  auto gtile = [&](const v16h& a0, const v16h& a1, const f16(*W)[3 * Ec], int tl) -> v8f {
    v16h b0 = make_frag_b(lane, [&](int k, int n) { return (float)W[k][tl * 16 + n]; });
    v16h b1 = make_frag_b(lane, [&](int k, int n) { return (float)W[32 + k][tl * 16 + n]; });
    v8f c = {};
    c = wmma(a0, b0, c);
    c = wmma(a1, b1, c);
    return c;
  };
  for (int t = 0; t < Tc; ++t) {
    size_t xbase = ((size_t)(b * Tc + t) * Nc + nbase);
    if (t + 1 < Tc)
      __builtin_prefetch(&x3[((size_t)(b * Tc + t + 1) * Nc + nbase) * Ec], 0, 1);
    v16h xa0 = make_frag_a(lane, [&](int m, int k) { return (float)x3[(xbase + m) * Ec + k]; });
    v16h xa1 = make_frag_a(lane, [&](int m, int k) { return (float)x3[(xbase + m) * Ec + 32 + k]; });
    v16h ha0 = make_frag_a(lane, [&](int m, int k) { return hS[w][m][k]; });
    v16h ha1 = make_frag_a(lane, [&](int m, int k) { return hS[w][m][32 + k]; });
#pragma unroll
    for (int tl = 0; tl < 4; ++tl) {
      v8f ir = gtile(xa0, xa1, WihT, tl);
      v8f iz = gtile(xa0, xa1, WihT, tl + 4);
      v8f ig = gtile(xa0, xa1, WihT, tl + 8);
      v8f hr = gtile(ha0, ha1, WhhT, tl);
      v8f hz = gtile(ha0, ha1, WhhT, tl + 4);
      v8f hg = gtile(ha0, ha1, WhhT, tl + 8);
      int cc = tl * 16 + (lane & 15);
#pragma unroll
      for (int i = 0; i < 8; ++i) {
        int m = i + 8 * ((lane >> 4) & 1);
        float r = sigm((ir[i] + bih[cc]) + (hr[i] + bhh[cc]));
        float z = sigm((iz[i] + bih[64 + cc]) + (hz[i] + bhh[64 + cc]));
        float nn = tanhf((ig[i] + bih[128 + cc]) + r * (hg[i] + bhh[128 + cc]));
        hS[w][m][cc] = (1.f - z) * nn + z * hS[w][m][cc];
      }
    }
    __syncthreads();
    if (lane < 16) {  // LNq + Wout -> qv
      int m = lane;
      float mu = 0.f;
      for (int c = 0; c < Ec; ++c) mu += hS[w][m][c];
      mu *= (1.f / Ec);
      float var = 0.f;
      for (int c = 0; c < Ec; ++c) { float d = hS[w][m][c] - mu; var += d * d; }
      float rs = rsqrtf(var * (1.f / Ec) + 1e-5f);
      float acc = 0.f;
      for (int c = 0; c < Ec; ++c)
        acc += ((hS[w][m][c] - mu) * rs * lnq_g[c] + lnq_b[c]) * Wout[c];
      qv[((size_t)b * Tc + t) * Nc + nbase + m] = acc + bo;
    }
    __syncthreads();
  }
  for (int idx = lane; idx < 16 * Ec; idx += 32) {
    int m = idx >> 6, c = idx & 63;
    hT[(size_t)(Rw + m) * Ec + c] = hS[w][m][c];
  }
}

extern "C" void kernel_launch(void* const* d_in, const int* in_sizes, int n_in,
                              void* d_out, int out_size, void* d_ws, size_t ws_size,
                              hipStream_t stream) {
  const float* states      = (const float*)d_in[0];
  const float* enemy_state = (const float*)d_in[1];
  const int*   actions     = (const int*)d_in[2];
  const float* rnn_h0      = (const float*)d_in[3];
  const float* agent_masks = (const float*)d_in[4];
  const float* gumbel      = (const float*)d_in[5];
  const float* agent_emb   = (const float*)d_in[6];
  const float* action_emb  = (const float*)d_in[7];
  const float* enemy_emb   = (const float*)d_in[8];
  const float* W_enemy = (const float*)d_in[9],  *b_enemy = (const float*)d_in[10];
  const float* W_ally  = (const float*)d_in[11], *b_ally  = (const float*)d_in[12];
  const float* ln1_g = (const float*)d_in[13], *ln1_b = (const float*)d_in[14];
  const float* Wk = (const float*)d_in[15], *bk = (const float*)d_in[16];
  const float* Wq = (const float*)d_in[17], *bq = (const float*)d_in[18];
  const float* Wv = (const float*)d_in[19], *bv = (const float*)d_in[20];
  const float* Wh = (const float*)d_in[21], *bh = (const float*)d_in[22];
  const float* ln2_g = (const float*)d_in[23], *ln2_b = (const float*)d_in[24];
  const float* Wl1 = (const float*)d_in[25], *bl1 = (const float*)d_in[26];
  const float* Wl2 = (const float*)d_in[27], *bl2 = (const float*)d_in[28];
  const float* ln3_g = (const float*)d_in[29], *ln3_b = (const float*)d_in[30];
  const float* Wc = (const float*)d_in[31], *bc = (const float*)d_in[32];
  const float* Wih = (const float*)d_in[33], *Whh = (const float*)d_in[34];
  const float* bih = (const float*)d_in[35], *bhh = (const float*)d_in[36];
  const float* lnq_g = (const float*)d_in[37], *lnq_b = (const float*)d_in[38];
  const float* Wout = (const float*)d_in[39], *bout = (const float*)d_in[40];
  (void)in_sizes; (void)n_in; (void)out_size; (void)ws_size;

  char* ws = (char*)d_ws;
  float* esum = (float*)(ws);                                   // 4096*64 f32
  float* se   = (float*)(ws + (size_t)1 * 1048576);             // 131072*64 f32
  float* oaeB = (float*)(ws + (size_t)1 * 1048576 + 33554432);  // 131072*64 f32
  f16*   qh   = (f16*)  (ws + (size_t)1 * 1048576 + 2 * 33554432ULL);
  f16*   kh   = (f16*)((char*)qh + 16777216);
  f16*   vh   = (f16*)((char*)kh + 16777216);
  float* agg  = se;        // se dead after projections
  f16*   x3   = qh;        // q dead after attention

  float* out = (float*)d_out;
  float* qv  = out;                                // B*T*N
  float* prd = out + (size_t)BTc * Nc;             // B*T*H*N*N
  float* hT  = prd + (size_t)BTc * Hc * Nc * Nc;   // B*N*E

  k_enemy<<<BTc, 32, 0, stream>>>(enemy_state, W_enemy, b_enemy, enemy_emb, esum);
  k_se<<<ROWS / 16, 32, 0, stream>>>(states, W_ally, b_ally, agent_emb, esum, ln1_g,
                                     ln1_b, actions, action_emb, se, oaeB);
  k_proj<<<ROWS / 16, 32, 0, stream>>>(se, Wq, bq, qh);
  k_proj<<<ROWS / 16, 32, 0, stream>>>(se, Wk, bk, kh);
  k_proj<<<ROWS / 16, 32, 0, stream>>>(oaeB, Wv, bv, vh);
  k_attn<<<BTc * Hc, 32, 0, stream>>>(qh, kh, vh, Wh, bh, gumbel, agent_masks, agg, prd);
  k_ffn<<<ROWS / 16, 32, 0, stream>>>(oaeB, agg, ln2_g, ln2_b, Wl1, bl1, Wl2, bl2,
                                      ln3_g, ln3_b, Wc, bc, x3);
  k_gru<<<32, 64, 0, stream>>>(x3, rnn_h0, Wih, Whh, bih, bhh, lnq_g, lnq_b, Wout,
                               bout, qv, hT);
}
